// GNN_29643864277577
// MI455X (gfx1250) — matrix operand order repaired and checked by reading, not compiled, not run on hardware
//
#include <hip/hip_runtime.h>

// ---------------------------------------------------------------------------
// Problem constants (match reference)
// ---------------------------------------------------------------------------
constexpr int   N_NODES  = 50000;
constexpr int   N_EDGES  = 600000;
constexpr int   D        = 128;
constexpr int   L        = 5;
constexpr int   N_GRAPHS = 256;
constexpr int   TOTAL_SUB = 2560;           // 256 * 10
constexpr float BN_EPS   = 1e-5f;
constexpr int   ND       = N_NODES * D;     // 6,400,000 (valid elements)
constexpr int   N_PAD    = 50016;           // rows padded to multiple of 32
constexpr int   ND_PAD   = N_PAD * D;

typedef __attribute__((ext_vector_type(16))) __bf16 v16bf;
typedef __attribute__((ext_vector_type(16))) float  v16f;
typedef __attribute__((ext_vector_type(8)))  float  v8f;

// Scalar float -> bf16 RNE via bits (cold paths only)
__device__ __forceinline__ __bf16 f2bf(float f) {
    unsigned u = __builtin_bit_cast(unsigned, f);
    unsigned r = (u + 0x7FFFu + ((u >> 16) & 1u)) >> 16;
    unsigned short s = (unsigned short)r;
    return __builtin_bit_cast(__bf16, s);
}

// ---------------------------------------------------------------------------
// Utility kernels
// ---------------------------------------------------------------------------
__global__ void k_zero(float* __restrict__ p, int n) {
    int i = blockIdx.x * blockDim.x + threadIdx.x;
    if (i < n) p[i] = 0.0f;
}

__global__ void k_copy4(const float4* __restrict__ s, float4* __restrict__ d, int n4) {
    int i = blockIdx.x * blockDim.x + threadIdx.x;
    if (i < n4) d[i] = s[i];
}

// Transpose weights to column-major bf16 (B-fragment = contiguous 16 K values
// of one column) and fuse the two bias vectors.
__global__ void k_prep(const float* __restrict__ Wr, const float* __restrict__ Wo,
                       const float* __restrict__ br, const float* __restrict__ bo,
                       __bf16* __restrict__ wtrel, __bf16* __restrict__ wtroot,
                       float* __restrict__ bias) {
    int i = blockIdx.x * blockDim.x + threadIdx.x;
    if (i < L * D * D) {
        int l = i / (D * D);
        int r = i % (D * D);
        int k = r / D, n = r % D;                         // W[l][k][n]
        size_t o = (size_t)l * D * D + (size_t)n * D + k; // Wt[l][n][k]
        wtrel[o]  = f2bf(Wr[i]);
        wtroot[o] = f2bf(Wo[i]);
    }
    if (i < L * D) bias[i] = br[i] + bo[i];
}

// Exclusive prefix over num_subgraphs (256 entries, trivial)
__global__ void k_offsets(const int* __restrict__ ns, int* __restrict__ offs) {
    if (blockIdx.x == 0 && threadIdx.x == 0) {
        int acc = 0;
        for (int g = 0; g < N_GRAPHS; ++g) { offs[g] = acc; acc += ns[g]; }
        offs[N_GRAPHS] = acc;
    }
}

// ---------------------------------------------------------------------------
// Edge scatter: agg[dst] += edge_attr * h[src]   (thread = edge x 4-feature)
// ---------------------------------------------------------------------------
__global__ void k_scatter(const int* __restrict__ ei, const float* __restrict__ ea,
                          const float* __restrict__ h, float* __restrict__ agg) {
    int t = blockIdx.x * blockDim.x + threadIdx.x;
    if (t >= N_EDGES * 32) return;
    int e = t >> 5, c = t & 31;
    int src = ei[e];
    int dst = ei[N_EDGES + e];
    float w = ea[e];
    float4 v = ((const float4*)h)[(size_t)src * 32 + c];
    float* o = agg + (size_t)dst * D + c * 4;
    atomicAdd(o + 0, v.x * w);
    atomicAdd(o + 1, v.y * w);
    atomicAdd(o + 2, v.z * w);
    atomicAdd(o + 3, v.w * w);
}

// ---------------------------------------------------------------------------
// WMMA GEMM: tmp = agg @ Wrel + h @ Wroot + bias, plus per-column sum/sumsq
// ---------------------------------------------------------------------------
// A-fragment (16x32 bf16): lane<16 -> M=lane, K = {0..15};
//                          lane>=16 -> M=lane-16, K = {16..31}
// Packed f32->bf16 conversion via convertvector (lets ISel emit v_cvt_pk).
__device__ __forceinline__ v16bf load_a(const float* __restrict__ p) {
    float4 f0 = *(const float4*)(p);
    float4 f1 = *(const float4*)(p + 4);
    float4 f2 = *(const float4*)(p + 8);
    float4 f3 = *(const float4*)(p + 12);
    v16f vf;
    vf[0]  = f0.x; vf[1]  = f0.y; vf[2]  = f0.z; vf[3]  = f0.w;
    vf[4]  = f1.x; vf[5]  = f1.y; vf[6]  = f1.z; vf[7]  = f1.w;
    vf[8]  = f2.x; vf[9]  = f2.y; vf[10] = f2.z; vf[11] = f2.w;
    vf[12] = f3.x; vf[13] = f3.y; vf[14] = f3.z; vf[15] = f3.w;
    return __builtin_convertvector(vf, v16bf);
}

__device__ __forceinline__ void store_tile(v8f acc, int col, int m0, int half,
                                           const float* __restrict__ bias,
                                           float* __restrict__ tmp,
                                           float* __restrict__ sums,
                                           float* __restrict__ sumsq) {
    const float bv  = bias[col];
    const int rbase = m0 + half * 8;      // C/D layout: lanes 16-31 hold M+8
    float s = 0.0f, ss = 0.0f;
    #pragma unroll
    for (int r = 0; r < 8; ++r) {
        float v = acc[r] + bv;
        tmp[(size_t)(rbase + r) * D + col] = v;
        s  += v;
        ss += v * v;
    }
    atomicAdd(&sums[col],  s);
    atomicAdd(&sumsq[col], ss);
}

__global__ __launch_bounds__(128)
void k_gemm(const float* __restrict__ agg, const float* __restrict__ h,
            const __bf16* __restrict__ wr, const __bf16* __restrict__ wo,
            const float* __restrict__ bias, float* __restrict__ tmp,
            float* __restrict__ sums, float* __restrict__ sumsq) {
    // Block covers 32 rows x 128 cols. Wave = 16 rows x 64 cols (4 tiles)
    // -> 4 independent accumulator chains to cover the WMMA->WMMA hazard.
    const int wave    = threadIdx.x >> 5;
    const int lane    = threadIdx.x & 31;
    const int m       = lane & 15;
    const int half    = lane >> 4;
    const int m0      = blockIdx.x * 32 + (wave >> 1) * 16;
    const int colHalf = wave & 1;                 // 0 -> cols 0..63, 1 -> 64..127
    const int arow    = m0 + m;
    const int kA      = half * 16;                // A K-offset per lane half
    const int kB      = half * 16;                // B K-offset per lane half
    const int colBase = colHalf * 64 + m;         // col of tile t is colBase + t*16

    v8f acc0 = {}, acc1 = {}, acc2 = {}, acc3 = {};
    const size_t arowOff = (size_t)arow * D;
    #pragma unroll
    for (int ks = 0; ks < 4; ++ks) {
        const int k0 = ks * 32;
        v16bf aA = load_a(agg + arowOff + k0 + kA);
        v16bf aH = load_a(h   + arowOff + k0 + kA);
        const size_t bo0 = (size_t)(colBase +  0) * D + k0 + kB;
        const size_t bo1 = (size_t)(colBase + 16) * D + k0 + kB;
        const size_t bo2 = (size_t)(colBase + 32) * D + k0 + kB;
        const size_t bo3 = (size_t)(colBase + 48) * D + k0 + kB;
        v16bf br0 = *(const v16bf*)(wr + bo0);
        v16bf br1 = *(const v16bf*)(wr + bo1);
        v16bf br2 = *(const v16bf*)(wr + bo2);
        v16bf br3 = *(const v16bf*)(wr + bo3);
        v16bf bo0v = *(const v16bf*)(wo + bo0);
        v16bf bo1v = *(const v16bf*)(wo + bo1);
        v16bf bo2v = *(const v16bf*)(wo + bo2);
        v16bf bo3v = *(const v16bf*)(wo + bo3);
        // Round-robin across 4 chains: dependent WMMAs are 4 apart.
        acc0 = __builtin_amdgcn_wmma_f32_16x16x32_bf16(false, aA, false, br0,  (short)0, acc0, false, false);
        acc1 = __builtin_amdgcn_wmma_f32_16x16x32_bf16(false, aA, false, br1,  (short)0, acc1, false, false);
        acc2 = __builtin_amdgcn_wmma_f32_16x16x32_bf16(false, aA, false, br2,  (short)0, acc2, false, false);
        acc3 = __builtin_amdgcn_wmma_f32_16x16x32_bf16(false, aA, false, br3,  (short)0, acc3, false, false);
        acc0 = __builtin_amdgcn_wmma_f32_16x16x32_bf16(false, aH, false, bo0v, (short)0, acc0, false, false);
        acc1 = __builtin_amdgcn_wmma_f32_16x16x32_bf16(false, aH, false, bo1v, (short)0, acc1, false, false);
        acc2 = __builtin_amdgcn_wmma_f32_16x16x32_bf16(false, aH, false, bo2v, (short)0, acc2, false, false);
        acc3 = __builtin_amdgcn_wmma_f32_16x16x32_bf16(false, aH, false, bo3v, (short)0, acc3, false, false);
    }
    if (m0 < N_NODES) {   // pad-row waves (last block) discard results
        store_tile(acc0, colBase +  0 - m + m, m0, half, bias, tmp, sums, sumsq);
        store_tile(acc1, colBase + 16, m0, half, bias, tmp, sums, sumsq);
        store_tile(acc2, colBase + 32, m0, half, bias, tmp, sums, sumsq);
        store_tile(acc3, colBase + 48, m0, half, bias, tmp, sums, sumsq);
    }
}

// ---------------------------------------------------------------------------
// BatchNorm stats finalize + apply, ReLU
// ---------------------------------------------------------------------------
__global__ void k_bnstats(const float* __restrict__ sums, const float* __restrict__ sumsq,
                          float* __restrict__ mu, float* __restrict__ rsig) {
    int i = threadIdx.x;
    if (i < D) {
        const float inv = 1.0f / (float)N_NODES;
        float m = sums[i] * inv;
        float v = sumsq[i] * inv - m * m;
        mu[i]   = m;
        rsig[i] = rsqrtf(v + BN_EPS);
    }
}

__global__ void k_bnapply(const float* __restrict__ tmp, const float* __restrict__ mu,
                          const float* __restrict__ rsig, const float* __restrict__ g,
                          const float* __restrict__ b, float* __restrict__ h) {
    int i = blockIdx.x * blockDim.x + threadIdx.x;   // over ND/4
    if (i >= ND / 4) return;
    int c = (i * 4) & (D - 1);                        // 4 consecutive cols, no wrap
    float4 v = ((const float4*)tmp)[i];
    float4 r;
    r.x = fmaxf(g[c + 0] * (v.x - mu[c + 0]) * rsig[c + 0] + b[c + 0], 0.0f);
    r.y = fmaxf(g[c + 1] * (v.y - mu[c + 1]) * rsig[c + 1] + b[c + 1], 0.0f);
    r.z = fmaxf(g[c + 2] * (v.z - mu[c + 2]) * rsig[c + 2] + b[c + 2], 0.0f);
    r.w = fmaxf(g[c + 3] * (v.w - mu[c + 3]) * rsig[c + 3] + b[c + 3], 0.0f);
    ((float4*)h)[i] = r;
}

// ---------------------------------------------------------------------------
// Subgraph mean pooling
// ---------------------------------------------------------------------------
__global__ void k_pool_scatter(const int* __restrict__ bat, const int* __restrict__ sg,
                               const int* __restrict__ offs, const float* __restrict__ h,
                               float* __restrict__ psums, float* __restrict__ pcnt) {
    int t = blockIdx.x * blockDim.x + threadIdx.x;
    if (t >= N_NODES * 32) return;
    int i = t >> 5, c = t & 31;
    int sid = sg[i] + offs[bat[i]];
    float4 v = ((const float4*)h)[(size_t)i * 32 + c];
    float* o = psums + (size_t)sid * D + c * 4;
    atomicAdd(o + 0, v.x);
    atomicAdd(o + 1, v.y);
    atomicAdd(o + 2, v.z);
    atomicAdd(o + 3, v.w);
    if (c == 0) atomicAdd(&pcnt[sid], 1.0f);
}

__global__ void k_pool_final(const float* __restrict__ psums, const float* __restrict__ pcnt,
                             float* __restrict__ out) {
    int i = blockIdx.x * blockDim.x + threadIdx.x;
    if (i >= TOTAL_SUB * D) return;
    int s = i >> 7;
    out[i] = psums[i] / fmaxf(pcnt[s], 1.0f);
}

// ---------------------------------------------------------------------------
// Host launcher
// ---------------------------------------------------------------------------
extern "C" void kernel_launch(void* const* d_in, const int* in_sizes, int n_in,
                              void* d_out, int out_size, void* d_ws, size_t ws_size,
                              hipStream_t stream) {
    const float* x        = (const float*)d_in[0];
    const float* eattr    = (const float*)d_in[1];
    const float* W_rel    = (const float*)d_in[2];
    const float* b_rel    = (const float*)d_in[3];
    const float* W_root   = (const float*)d_in[4];
    const float* b_root   = (const float*)d_in[5];
    const float* gamma    = (const float*)d_in[6];
    const float* beta     = (const float*)d_in[7];
    const int*   eindex   = (const int*)d_in[8];
    const int*   batch    = (const int*)d_in[9];
    const int*   num_sub  = (const int*)d_in[10];
    const int*   sg_batch = (const int*)d_in[11];
    float*       out      = (float*)d_out;

    char* base = (char*)d_ws;
    size_t o = 0;
    float*  agg    = (float*)(base + o); o += (size_t)ND_PAD * 4;        // ~25.6 MB
    float*  tmp    = (float*)(base + o); o += (size_t)ND_PAD * 4;        // ~25.6 MB
    float*  hbuf   = (float*)(base + o); o += (size_t)ND_PAD * 4;        // ~25.6 MB
    __bf16* wtrel  = (__bf16*)(base + o); o += (size_t)L * D * D * 2;    // 160 KB
    __bf16* wtroot = (__bf16*)(base + o); o += (size_t)L * D * D * 2;    // 160 KB
    float*  bias   = (float*)(base + o); o += (size_t)L * D * 4;
    float*  sums   = (float*)(base + o); o += 512;   // contiguous with sumsq
    float*  sumsq  = (float*)(base + o); o += 512;
    float*  mu     = (float*)(base + o); o += 512;
    float*  rsig   = (float*)(base + o); o += 512;
    int*    offs   = (int*)(base + o);   o += (N_GRAPHS + 1) * 4;
    o = (o + 255) & ~(size_t)255;
    float*  psums  = (float*)(base + o); o += (size_t)TOTAL_SUB * D * 4; // contiguous
    float*  pcnt   = (float*)(base + o); o += (size_t)TOTAL_SUB * 4;     // with psums
    (void)ws_size; (void)in_sizes; (void)n_in; (void)out_size;

    const int B = 256;

    // One-time-per-call prep: bf16 transposed weights, fused bias, offsets, h=x
    k_prep<<<(L * D * D + B - 1) / B, B, 0, stream>>>(W_rel, W_root, b_rel, b_root,
                                                      wtrel, wtroot, bias);
    k_offsets<<<1, 1, 0, stream>>>(num_sub, offs);
    k_copy4<<<(ND / 4 + B - 1) / B, B, 0, stream>>>((const float4*)x, (float4*)hbuf, ND / 4);

    for (int l = 0; l < L; ++l) {
        k_zero<<<(ND + B - 1) / B, B, 0, stream>>>(agg, ND);
        k_zero<<<1, B, 0, stream>>>(sums, 2 * D);   // sums + sumsq (contiguous)
        k_scatter<<<(N_EDGES * 32) / B, B, 0, stream>>>(eindex, eattr, hbuf, agg);
        k_gemm<<<N_PAD / 32, 128, 0, stream>>>(agg, hbuf,
                                               wtrel + (size_t)l * D * D,
                                               wtroot + (size_t)l * D * D,
                                               bias + l * D, tmp, sums, sumsq);
        k_bnstats<<<1, 128, 0, stream>>>(sums, sumsq, mu, rsig);
        k_bnapply<<<(ND / 4 + B - 1) / B, B, 0, stream>>>(tmp, mu, rsig,
                                                          gamma + l * D, beta + l * D, hbuf);
    }

    k_zero<<<(TOTAL_SUB * D + TOTAL_SUB + B - 1) / B, B, 0, stream>>>(psums,
                                                                     TOTAL_SUB * D + TOTAL_SUB);
    k_pool_scatter<<<(N_NODES * 32 + B - 1) / B, B, 0, stream>>>(batch, sg_batch, offs,
                                                                 hbuf, psums, pcnt);
    k_pool_final<<<(TOTAL_SUB * D + B - 1) / B, B, 0, stream>>>(psums, pcnt, out);
}
